// MultiHeadAttention_64819646431772
// MI455X (gfx1250) — compile-verified, matrix operand order checked by reading, not run on hardware
//
#include <hip/hip_runtime.h>

#define BDIM 2
#define SDIM 2048
#define DDIM 1024
#define HDIM 16
#define HDD  64

typedef __attribute__((ext_vector_type(16))) __bf16          v16bf;
typedef __attribute__((ext_vector_type(16))) unsigned short  v16us;
typedef __attribute__((ext_vector_type(8)))  float           v8f;

__device__ __forceinline__ unsigned short f32_to_bf16(float f) {
  union { float f; unsigned u; } c; c.f = f;
  unsigned x = c.u;
  return (unsigned short)((x + 0x7FFFu + ((x >> 16) & 1u)) >> 16);  // RNE
}

// k index (base of a contiguous pair) held by element-pair p of a 16-bit A/B
// fragment, per CDNA5 ISA 7.12.2: lanes 0-15 hold K[0..7]+[16..23],
// lanes 16-31 hold K[8..15]+[24..31], 2 halves per VGPR.
__device__ __forceinline__ int frag_k(int lane, int p) {
  return ((p & 3) << 1) + ((lane & 16) ? 8 : 0) + ((p >> 2) << 4);
}

// Fragment where the K dimension is contiguous in memory. Lowers to
// 2x b128 loads per lane (verified in round-1 disasm).
__device__ __forceinline__ v16us load_frag_kcontig(const unsigned short* base,
                                                   int ld, int lane) {
  int r = lane & 15;
  v16us v;
#pragma unroll
  for (int p = 0; p < 8; ++p) {
    int k = frag_k(lane, p);
    const unsigned short* s = base + (size_t)r * ld + k;
    v[2 * p]     = s[0];
    v[2 * p + 1] = s[1];
  }
  return v;
}

__device__ __forceinline__ v8f wmma_bf16(v16us a, v16us b, v8f c) {
  return __builtin_amdgcn_wmma_f32_16x16x32_bf16(
      false, __builtin_bit_cast(v16bf, a),
      false, __builtin_bit_cast(v16bf, b),
      (short)0, c, false, false);
}

__global__ void cvt_bf16_kernel(const float* __restrict__ in,
                                unsigned short* __restrict__ out, int n) {
  int i = blockIdx.x * blockDim.x + threadIdx.x;
  if (i < n) out[i] = f32_to_bf16(in[i]);
}

// Q/K/V projection: out = x @ W^T + b.
// Q,K stored bf16 head-split [B,H,S,HD]; V stored TRANSPOSED [B,H,HD,S] so the
// attention P@V B-fragments are K-contiguous (wide loads, no strided gather).
// Block: 256 threads = 8 waves; each wave register-blocks a 32x32 output tile
// (2x2 WMMA tiles -> 4 WMMAs per 4 fragment loads).
__global__ void qkv_proj_kernel(const unsigned short* __restrict__ xb,
                                const unsigned short* __restrict__ Wqb,
                                const unsigned short* __restrict__ Wkb,
                                const unsigned short* __restrict__ Wvb,
                                const float* __restrict__ bq,
                                const float* __restrict__ bk,
                                const float* __restrict__ bv,
                                unsigned short* __restrict__ Qh,
                                unsigned short* __restrict__ Kh,
                                unsigned short* __restrict__ Vt) {
  const unsigned short* W = (blockIdx.z == 0) ? Wqb : (blockIdx.z == 1) ? Wkb : Wvb;
  const float* bias       = (blockIdx.z == 0) ? bq  : (blockIdx.z == 1) ? bk  : bv;
  unsigned short* outh    = (blockIdx.z == 0) ? Qh  : (blockIdx.z == 1) ? Kh  : Vt;

  int w = threadIdx.x >> 5, lane = threadIdx.x & 31;
  int m0 = blockIdx.x * 64 + (w & 1) * 32;    // 2 row tiles: m0, m0+16
  int n0 = blockIdx.y * 128 + (w >> 1) * 32;  // 2 col tiles: n0, n0+16

  v8f acc[2][2] = {};
  for (int k0 = 0; k0 < DDIM; k0 += 32) {
    __builtin_prefetch(W + (size_t)n0 * DDIM + k0 + 256, 0, 1);
    v16us a0 = load_frag_kcontig(xb + (size_t)m0 * DDIM + k0, DDIM, lane);
    v16us a1 = load_frag_kcontig(xb + (size_t)(m0 + 16) * DDIM + k0, DDIM, lane);
    v16us b0 = load_frag_kcontig(W + (size_t)n0 * DDIM + k0, DDIM, lane);
    v16us b1 = load_frag_kcontig(W + (size_t)(n0 + 16) * DDIM + k0, DDIM, lane);
    acc[0][0] = wmma_bf16(a0, b0, acc[0][0]);
    acc[0][1] = wmma_bf16(a0, b1, acc[0][1]);
    acc[1][0] = wmma_bf16(a1, b0, acc[1][0]);
    acc[1][1] = wmma_bf16(a1, b1, acc[1][1]);
  }
#pragma unroll
  for (int ti = 0; ti < 2; ++ti)
#pragma unroll
    for (int tj = 0; tj < 2; ++tj)
#pragma unroll
      for (int g = 0; g < 8; ++g) {
        int Mr = m0 + ti * 16 + g + ((lane >> 4) << 3);
        int Nc = n0 + tj * 16 + (lane & 15);
        float val = acc[ti][tj][g] + bias[Nc];
        int bI = Mr >> 11;            // row -> batch (S = 2048)
        int s  = Mr & (SDIM - 1);
        int hh = Nc >> 6;             // col -> head (HD = 64)
        int hd = Nc & 63;
        unsigned short bv16 = f32_to_bf16(val);
        if (blockIdx.z == 2)          // V: [B,H,HD,S] (transposed)
          outh[(((size_t)bI * HDIM + hh) * HDD + hd) * SDIM + s] = bv16;
        else                          // Q,K: [B,H,S,HD]
          outh[(((size_t)bI * HDIM + hh) * SDIM + s) * HDD + hd] = bv16;
      }
}

// Fused attention per (batch, 16-query-row tile), looping all heads.
// LDS: scores f32 [16][2048] (128KB, reused as P@V partial buffer)
//      + probs bf16 [16][2048] (64KB) + Q tile bf16 [16][64] (2KB) = 194KB.
__global__ void attn_kernel(const unsigned short* __restrict__ Qh,
                            const unsigned short* __restrict__ Kh,
                            const unsigned short* __restrict__ Vt,
                            unsigned short* __restrict__ ctxb,
                            float* __restrict__ avg) {
  extern __shared__ char smem[];
  float*          sc = (float*)smem;                                      // 16x2048 f32
  unsigned short* pr = (unsigned short*)(smem + 16 * SDIM * 4);           // 16x2048 bf16
  unsigned short* qt = (unsigned short*)(smem + 16 * SDIM * 4 + 16 * SDIM * 2);

  int q0 = blockIdx.x * 16;
  int bI = blockIdx.y;
  int w = threadIdx.x >> 5, lane = threadIdx.x & 31;

  for (int h = 0; h < HDIM; ++h) {
    const unsigned short* Qbh = Qh + (((size_t)bI * HDIM + h) * SDIM) * HDD;
    const unsigned short* Kbh = Kh + (((size_t)bI * HDIM + h) * SDIM) * HDD;
    const unsigned short* Vbh = Vt + (((size_t)bI * HDIM + h) * HDD) * SDIM;

    // stage Q tile
    for (int i = threadIdx.x; i < 16 * HDD; i += 256)
      qt[i] = Qbh[(size_t)(q0 + (i >> 6)) * HDD + (i & 63)];
    __syncthreads();

    // Q-tile A fragments are invariant across all key tiles: load once.
    v16us aq0 = load_frag_kcontig(qt + 0,  HDD, lane);
    v16us aq1 = load_frag_kcontig(qt + 32, HDD, lane);

    // scores[16][2048] = (Q_tile @ K^T) / sqrt(HD); 8 waves x 16 key tiles
    for (int nt = w; nt < SDIM / 16; nt += 8) {
      const unsigned short* kb = Kbh + (size_t)(nt * 16) * HDD;
      __builtin_prefetch(kb + 8 * 16 * HDD, 0, 1);
      v16us b0 = load_frag_kcontig(kb + 0,  HDD, lane);
      v16us b1 = load_frag_kcontig(kb + 32, HDD, lane);
      v8f acc = {};
      acc = wmma_bf16(aq0, b0, acc);
      acc = wmma_bf16(aq1, b1, acc);
#pragma unroll
      for (int g = 0; g < 8; ++g) {
        int Mr = g + ((lane >> 4) << 3);
        int Nc = nt * 16 + (lane & 15);
        sc[(size_t)Mr * SDIM + Nc] = acc[g] * 0.125f;  // 1/sqrt(64)
      }
    }
    __syncthreads();

    // softmax: each wave handles 2 rows; fold into avg_attn (exclusive row
    // ownership -> plain global RMW, stays hot in the 192MB L2)
#pragma unroll
    for (int rr = 0; rr < 2; ++rr) {
      int r = w * 2 + rr;
      float* row = sc + (size_t)r * SDIM;
      float m = -3.4e38f;
      for (int j = lane; j < SDIM; j += 32) m = fmaxf(m, row[j]);
#pragma unroll
      for (int d = 16; d > 0; d >>= 1) m = fmaxf(m, __shfl_xor(m, d, 32));
      float ssum = 0.f;
      for (int j = lane; j < SDIM; j += 32) {
        float e = __expf(row[j] - m);
        row[j] = e;
        ssum += e;
      }
#pragma unroll
      for (int d = 16; d > 0; d >>= 1) ssum += __shfl_xor(ssum, d, 32);
      float inv = 1.0f / ssum;
      float* ag = avg + ((size_t)bI * SDIM + q0 + r) * SDIM;
      unsigned short* pro = pr + (size_t)r * SDIM;
      for (int j = lane; j < SDIM; j += 32) {
        float p = row[j] * inv;
        pro[j] = f32_to_bf16(p);
        float contrib = p * (1.0f / HDIM);
        if (h == 0) ag[j] = contrib; else ag[j] += contrib;
      }
    }
    __syncthreads();

    // ctx tile = P @ V. 8 waves: wave w does tile nt=w&3 over half the keys
    // (kh=w>>2), partials combined through LDS (sc is free now).
    {
      int nt = w & 3;
      int kh = w >> 2;
      const unsigned short* vtb = Vbh + (size_t)(nt * 16) * SDIM;  // rows=hd, ld=S
      v8f acc = {};
      for (int kc = kh * 32; kc < kh * 32 + 32; ++kc) {
        v16us a = load_frag_kcontig(pr + kc * 32, SDIM, lane);
        v16us b = load_frag_kcontig(vtb + kc * 32, SDIM, lane);
        acc = wmma_bf16(a, b, acc);
      }
      float* part = sc + (size_t)w * 256;
#pragma unroll
      for (int g = 0; g < 8; ++g) part[lane * 8 + g] = acc[g];
    }
    __syncthreads();

    // combine the two K-halves and write ctx in combined-head layout
    for (int i = threadIdx.x; i < 4 * 256; i += 256) {
      int nt = i >> 8;
      int e  = i & 255;
      float vsum = sc[nt * 256 + e] + sc[(nt + 4) * 256 + e];
      int lane_e = e >> 3, g = e & 7;
      int Mr = q0 + g + ((lane_e >> 4) << 3);
      int Nc = h * HDD + nt * 16 + (lane_e & 15);
      ctxb[((size_t)bI * SDIM + Mr) * DDIM + Nc] = f32_to_bf16(vsum);
    }
    __syncthreads();
  }
}

// out = ctx @ Wo^T + bo  (f32 output), 2x2 register-blocked per wave.
__global__ void out_proj_kernel(const unsigned short* __restrict__ ctxb,
                                const unsigned short* __restrict__ Wob,
                                const float* __restrict__ bo,
                                float* __restrict__ out) {
  int w = threadIdx.x >> 5, lane = threadIdx.x & 31;
  int m0 = blockIdx.x * 64 + (w & 1) * 32;
  int n0 = blockIdx.y * 128 + (w >> 1) * 32;
  v8f acc[2][2] = {};
  for (int k0 = 0; k0 < DDIM; k0 += 32) {
    __builtin_prefetch(Wob + (size_t)n0 * DDIM + k0 + 256, 0, 1);
    v16us a0 = load_frag_kcontig(ctxb + (size_t)m0 * DDIM + k0, DDIM, lane);
    v16us a1 = load_frag_kcontig(ctxb + (size_t)(m0 + 16) * DDIM + k0, DDIM, lane);
    v16us b0 = load_frag_kcontig(Wob + (size_t)n0 * DDIM + k0, DDIM, lane);
    v16us b1 = load_frag_kcontig(Wob + (size_t)(n0 + 16) * DDIM + k0, DDIM, lane);
    acc[0][0] = wmma_bf16(a0, b0, acc[0][0]);
    acc[0][1] = wmma_bf16(a0, b1, acc[0][1]);
    acc[1][0] = wmma_bf16(a1, b0, acc[1][0]);
    acc[1][1] = wmma_bf16(a1, b1, acc[1][1]);
  }
#pragma unroll
  for (int ti = 0; ti < 2; ++ti)
#pragma unroll
    for (int tj = 0; tj < 2; ++tj)
#pragma unroll
      for (int g = 0; g < 8; ++g) {
        int Mr = m0 + ti * 16 + g + ((lane >> 4) << 3);
        int Nc = n0 + tj * 16 + (lane & 15);
        out[(size_t)Mr * DDIM + Nc] = acc[ti][tj][g] + bo[Nc];
      }
}

extern "C" void kernel_launch(void* const* d_in, const int* in_sizes, int n_in,
                              void* d_out, int out_size, void* d_ws, size_t ws_size,
                              hipStream_t stream) {
  const float* x  = (const float*)d_in[0];
  const float* Wq = (const float*)d_in[1];
  const float* bq = (const float*)d_in[2];
  const float* Wk = (const float*)d_in[3];
  const float* bk = (const float*)d_in[4];
  const float* Wv = (const float*)d_in[5];
  const float* bv = (const float*)d_in[6];
  const float* Wo = (const float*)d_in[7];
  const float* bo = (const float*)d_in[8];

  float* out = (float*)d_out;
  float* avg = out + (size_t)BDIM * SDIM * DDIM;

  const size_t MD = (size_t)BDIM * SDIM * DDIM;  // 4Mi elements
  const size_t DD = (size_t)DDIM * DDIM;         // 1Mi elements
  unsigned short* xb   = (unsigned short*)d_ws;  // total ws use: 48 MiB bf16
  unsigned short* wqb  = xb + MD;
  unsigned short* wkb  = wqb + DD;
  unsigned short* wvb  = wkb + DD;
  unsigned short* wob  = wvb + DD;
  unsigned short* Qh   = wob + DD;
  unsigned short* Kh   = Qh + MD;
  unsigned short* Vt   = Kh + MD;
  unsigned short* ctxb = Vt + MD;

  const int T = 256;
  cvt_bf16_kernel<<<(int)((MD + T - 1) / T), T, 0, stream>>>(x,  xb,  (int)MD);
  cvt_bf16_kernel<<<(int)((DD + T - 1) / T), T, 0, stream>>>(Wq, wqb, (int)DD);
  cvt_bf16_kernel<<<(int)((DD + T - 1) / T), T, 0, stream>>>(Wk, wkb, (int)DD);
  cvt_bf16_kernel<<<(int)((DD + T - 1) / T), T, 0, stream>>>(Wv, wvb, (int)DD);
  cvt_bf16_kernel<<<(int)((DD + T - 1) / T), T, 0, stream>>>(Wo, wob, (int)DD);

  qkv_proj_kernel<<<dim3(BDIM * SDIM / 64, DDIM / 128, 3), T, 0, stream>>>(
      xb, wqb, wkb, wvb, bq, bk, bv, Qh, Kh, Vt);

  size_t smem = (size_t)16 * SDIM * 4 + (size_t)16 * SDIM * 2 + (size_t)16 * HDD * 2;
  (void)hipFuncSetAttribute((const void*)attn_kernel,
                            hipFuncAttributeMaxDynamicSharedMemorySize,
                            (int)smem);
  attn_kernel<<<dim3(SDIM / 16, BDIM), T, smem, stream>>>(Qh, Kh, Vt, ctxb, avg);

  out_proj_kernel<<<dim3(BDIM * SDIM / 64, DDIM / 128), T, 0, stream>>>(
      ctxb, wob, bo, out);
}
